// MultiHeadSelfAttention_17008070492176
// MI455X (gfx1250) — matrix operand order repaired and checked by reading, not compiled, not run on hardware
//
#include <hip/hip_runtime.h>
#include <hip/hip_bf16.h>

// MI455X (gfx1250) multi-head self-attention.
// bf16 WMMA everywhere + flash-attention; GEMMs stage tiles to LDS with
// GLOBAL_LOAD_ASYNC_TO_LDS_B128 (ASYNCcnt double-buffer pipeline).
// B=2, T=2048, D=1024, H=16, Dh=64.

typedef __attribute__((ext_vector_type(16))) __bf16 v16bf;
typedef __attribute__((ext_vector_type(8)))  float  v8f;
typedef __attribute__((__vector_size__(16))) int    v4i;   // matches builtin param

#define WMMA_BF16(A, Bm, C) \
    __builtin_amdgcn_wmma_f32_16x16x32_bf16(false, (A), false, (Bm), (short)0, (C), false, false)

__device__ __forceinline__ uint4 ld4(const __bf16* p) { return *(const uint4*)p; }

__device__ __forceinline__ v16bf mkf(uint4 lo, uint4 hi) {
    union { uint4 u[2]; v16bf v; } c;
    c.u[0] = lo; c.u[1] = hi;
    return c.v;
}

// ---- CDNA5 async copy: 16B per lane, global -> LDS, tracked by ASYNCcnt ----
__device__ __forceinline__ void g2l_b128(const __bf16* g, __bf16* l) {
#if __has_builtin(__builtin_amdgcn_global_load_async_to_lds_b128)
    void* gv = (void*)g;   // drop const
    __builtin_amdgcn_global_load_async_to_lds_b128(
        (__attribute__((address_space(1))) v4i*)gv,
        (__attribute__((address_space(3))) v4i*)l, 0, 0);
#else
    const unsigned loff = (unsigned)(uintptr_t)(__attribute__((address_space(3))) void*)l;
    asm volatile("global_load_async_to_lds_b128 %0, %1, off"
                 :: "v"(loff), "v"((unsigned long long)(uintptr_t)g) : "memory");
#endif
}
template <int N>
__device__ __forceinline__ void wait_async_le() {
    asm volatile("s_wait_asynccnt %0" :: "i"(N) : "memory");
}
__device__ __forceinline__ void wait_ds0() {
    asm volatile("s_wait_dscnt 0x0" ::: "memory");
}

// ---- kernel 0: fp32 -> bf16 pre-conversion (8 floats / thread) -------------
__global__ void cvt_bf16_kernel(const float* __restrict__ s, __bf16* __restrict__ d) {
    const int i = blockIdx.x * blockDim.x + threadIdx.x;
    const float4* sp = (const float4*)s + (size_t)i * 2;
    const float4 f0 = sp[0], f1 = sp[1];
    union { __bf16 b[8]; uint4 u; } o;
    o.b[0] = (__bf16)f0.x; o.b[1] = (__bf16)f0.y;
    o.b[2] = (__bf16)f0.z; o.b[3] = (__bf16)f0.w;
    o.b[4] = (__bf16)f1.x; o.b[5] = (__bf16)f1.y;
    o.b[6] = (__bf16)f1.z; o.b[7] = (__bf16)f1.w;
    ((uint4*)d)[i] = o.u;
}

// Per-wave LDS tile: A 16x32 bf16 (1KB) then B 64x32 bf16 (4KB) = 2560 elems.
#define TILE_ELEMS 2560
#define B_OFF 512

// Issue one k-step tile (10 async b128 ops: 2 for A, 8 for B).
__device__ __forceinline__ void issue_tile(const __bf16* gA0, const __bf16* gB0,
                                           __bf16* lbuf, int k0, int r4, int s4) {
    // A: 16 rows x 32 k ; one op covers 8 rows (4 lanes/row, 8 elems/lane)
    g2l_b128(gA0 + k0,            lbuf + r4 * 32 + s4 * 8);
    g2l_b128(gA0 + k0 + 8 * 1024, lbuf + (8 + r4) * 32 + s4 * 8);
    // B: 64 rows x 32 k
#pragma unroll
    for (int j = 0; j < 8; ++j)
        g2l_b128(gB0 + (size_t)j * 8 * 1024 + k0,
                 lbuf + B_OFF + (j * 8 + r4) * 32 + s4 * 8);
}

// ---- kernel 1: QKV projection [4096,1024] x Wqkv^T[1024,3072] --------------
// Fragment layouts used here (contiguous k, not the ISA table order):
//   A lane L elem e -> M=L%16, K=16*(L/16)+e
//   B lane L elem e -> K=16*(L/16)+e, N=L%16
__global__ void gemm_qkv_kernel(const __bf16* __restrict__ xbf,
                                const __bf16* __restrict__ Wbf,
                                __bf16* __restrict__ Qw,
                                __bf16* __restrict__ Kw,
                                __bf16* __restrict__ Vtw) {
    __shared__ __bf16 lds[4][2][TILE_ELEMS];
    const int wave = blockIdx.x * (blockDim.x >> 5) + (threadIdx.x >> 5);
    const int nt = wave % 48;   // 3072/64
    const int mt = wave / 48;   // 4096/16
    const int lane = threadIdx.x & 31;
    const int mloc = lane & 15, half = lane >> 4;
    const int w = threadIdx.x >> 5;
    const int r4 = lane >> 2, s4 = lane & 3;

    const __bf16* gA0 = xbf + (size_t)(mt * 16 + r4) * 1024 + s4 * 8;
    const __bf16* gB0 = Wbf + (size_t)(nt * 64 + r4) * 1024 + s4 * 8;
    __bf16* buf0 = &lds[w][0][0];
    __bf16* buf1 = &lds[w][1][0];

    issue_tile(gA0, gB0, buf0, 0,  r4, s4);
    issue_tile(gA0, gB0, buf1, 32, r4, s4);

    v8f acc0 = {}, acc1 = {}, acc2 = {}, acc3 = {};
    for (int step = 0; step < 32; ++step) {
        wait_async_le<10>();                    // tile `step` resident in LDS
        __bf16* lb = (step & 1) ? buf1 : buf0;
        const __bf16* Ab = lb;
        const __bf16* Bb = lb + B_OFF;

        const v16bf a = mkf(ld4(Ab + mloc * 32 + 16 * half),
                            ld4(Ab + mloc * 32 + 16 * half + 8));
        const v16bf b0 = mkf(ld4(Bb + (0  + mloc) * 32 + 16 * half),
                             ld4(Bb + (0  + mloc) * 32 + 16 * half + 8));
        const v16bf b1 = mkf(ld4(Bb + (16 + mloc) * 32 + 16 * half),
                             ld4(Bb + (16 + mloc) * 32 + 16 * half + 8));
        const v16bf b2 = mkf(ld4(Bb + (32 + mloc) * 32 + 16 * half),
                             ld4(Bb + (32 + mloc) * 32 + 16 * half + 8));
        const v16bf b3 = mkf(ld4(Bb + (48 + mloc) * 32 + 16 * half),
                             ld4(Bb + (48 + mloc) * 32 + 16 * half + 8));
        acc0 = WMMA_BF16(a, b0, acc0);
        acc1 = WMMA_BF16(a, b1, acc1);
        acc2 = WMMA_BF16(a, b2, acc2);
        acc3 = WMMA_BF16(a, b3, acc3);

        wait_ds0();                             // LDS reads done -> buffer reusable
        issue_tile(gA0, gB0, lb, ((step + 2) & 31) * 32, r4, s4);
    }

    const int sec = (nt * 64) >> 10;            // 0=Q,1=K,2=V
    v8f accs[4] = {acc0, acc1, acc2, acc3};
#pragma unroll
    for (int i = 0; i < 4; ++i) {
#pragma unroll
        for (int r = 0; r < 8; ++r) {
            const int m = mt * 16 + r + 8 * half;
            const int b = m >> 11, t = m & 2047;
            const int n = nt * 64 + 16 * i + mloc;
            const int dfull = n & 1023;
            const int h = dfull >> 6, d = dfull & 63;
            const size_t bh = (size_t)(b * 16 + h);
            const float v = accs[i][r];
            if (sec == 0)       Qw[(bh * 2048 + t) * 64 + d] = (__bf16)(v * 0.125f);
            else if (sec == 1)  Kw[(bh * 2048 + t) * 64 + d] = (__bf16)v;
            else                Vtw[(bh * 64 + d) * 2048 + t] = (__bf16)v;
        }
    }
}

// ---- kernel 2: flash attention, one wave per (b,h, 16 query rows) ----------
// Register double-buffer: prefetch next 32 kv-columns' K/V fragments so a full
// softmax + 8 WMMA body covers their latency.
__global__ void attn_kernel(const __bf16* __restrict__ Qw,
                            const __bf16* __restrict__ Kw,
                            const __bf16* __restrict__ Vtw,
                            const unsigned char* __restrict__ mask,
                            __bf16* __restrict__ Aout) {
    const int wave = blockIdx.x * (blockDim.x >> 5) + (threadIdx.x >> 5);
    const int qt = wave & 127;          // T/16
    const int bh = wave >> 7;           // 0..31
    const int b = bh >> 4, h = bh & 15;
    const int lane = threadIdx.x & 31;
    const int mloc = lane & 15, half = lane >> 4;
    const int w = threadIdx.x >> 5;

    const __bf16* Qbh = Qw + (size_t)bh * 2048 * 64;
    const __bf16* Kbh = Kw + (size_t)bh * 2048 * 64;
    const __bf16* Vbh = Vtw + (size_t)bh * 64 * 2048;
    const unsigned char* mrow = mask + (size_t)b * 2048;

    __shared__ __bf16 Plds[4][16][32];

    const __bf16* qrow = Qbh + (size_t)(qt * 16 + mloc) * 64;
    const v16bf qa0 = mkf(ld4(qrow + 16 * half),      ld4(qrow + 8 + 16 * half));
    const v16bf qa1 = mkf(ld4(qrow + 32 + 16 * half), ld4(qrow + 40 + 16 * half));

    const __bf16* kpb = Kbh + (size_t)mloc * 64 + 16 * half;   // + t*64
    const __bf16* vr0 = Vbh + (size_t)(0  + mloc) * 2048 + 16 * half;
    const __bf16* vr1 = Vbh + (size_t)(16 + mloc) * 2048 + 16 * half;
    const __bf16* vr2 = Vbh + (size_t)(32 + mloc) * 2048 + 16 * half;
    const __bf16* vr3 = Vbh + (size_t)(48 + mloc) * 2048 + 16 * half;

    v8f o0 = {}, o1 = {}, o2 = {}, o3 = {};
    float mr[8], lr[8];
#pragma unroll
    for (int r = 0; r < 8; ++r) { mr[r] = -3.0e38f; lr[r] = 0.0f; }

    // prologue: load fragments for t0 = 0
    uint4 ka0l = ld4(kpb), ka0h = ld4(kpb + 8);
    uint4 kb0l = ld4(kpb + 32), kb0h = ld4(kpb + 40);
    uint4 ka1l = ld4(kpb + 16 * 64), ka1h = ld4(kpb + 16 * 64 + 8);
    uint4 kb1l = ld4(kpb + 16 * 64 + 32), kb1h = ld4(kpb + 16 * 64 + 40);
    uint4 v0l = ld4(vr0), v0h = ld4(vr0 + 8);
    uint4 v1l = ld4(vr1), v1h = ld4(vr1 + 8);
    uint4 v2l = ld4(vr2), v2h = ld4(vr2 + 8);
    uint4 v3l = ld4(vr3), v3h = ld4(vr3 + 8);
    unsigned char km0b = mrow[mloc], km1b = mrow[16 + mloc];

    for (int t0 = 0; t0 < 2048; t0 += 32) {
        const int tn = (t0 + 32) & 2047;        // wrapped prefetch index
        const __bf16* kn0 = kpb + (size_t)tn * 64;
        const __bf16* kn1 = kn0 + 16 * 64;
        uint4 nka0l = ld4(kn0), nka0h = ld4(kn0 + 8);
        uint4 nkb0l = ld4(kn0 + 32), nkb0h = ld4(kn0 + 40);
        uint4 nka1l = ld4(kn1), nka1h = ld4(kn1 + 8);
        uint4 nkb1l = ld4(kn1 + 32), nkb1h = ld4(kn1 + 40);
        uint4 nv0l = ld4(vr0 + tn), nv0h = ld4(vr0 + tn + 8);
        uint4 nv1l = ld4(vr1 + tn), nv1h = ld4(vr1 + tn + 8);
        uint4 nv2l = ld4(vr2 + tn), nv2h = ld4(vr2 + tn + 8);
        uint4 nv3l = ld4(vr3 + tn), nv3h = ld4(vr3 + tn + 8);
        unsigned char nkm0 = mrow[tn + mloc], nkm1 = mrow[tn + 16 + mloc];

        // S = (Q*scale) K^T : two 16x16 f32 tiles
        v8f s0 = {}, s1 = {};
        s0 = WMMA_BF16(qa0, mkf(ka0l, ka0h), s0);
        s0 = WMMA_BF16(qa1, mkf(kb0l, kb0h), s0);
        s1 = WMMA_BF16(qa0, mkf(ka1l, ka1h), s1);
        s1 = WMMA_BF16(qa1, mkf(kb1l, kb1h), s1);
        const bool km0 = km0b != 0, km1 = km1b != 0;

        // online softmax; a row spans the 16 lanes of one half-group
#pragma unroll
        for (int r = 0; r < 8; ++r) {
            float a = km0 ? -3.0e38f : s0[r];
            float c = km1 ? -3.0e38f : s1[r];
            float mx = fmaxf(a, c);
#pragma unroll
            for (int off = 8; off >= 1; off >>= 1)
                mx = fmaxf(mx, __shfl_xor(mx, off, 32));
            const float mnew = fmaxf(mr[r], mx);
            const float fac = __expf(mr[r] - mnew);
            const float p0 = __expf(a - mnew);
            const float p1 = __expf(c - mnew);
            float rs = p0 + p1;
#pragma unroll
            for (int off = 8; off >= 1; off >>= 1)
                rs += __shfl_xor(rs, off, 32);
            lr[r] = lr[r] * fac + rs;
            mr[r] = mnew;
            o0[r] *= fac; o1[r] *= fac; o2[r] *= fac; o3[r] *= fac;
            Plds[w][r + 8 * half][mloc]      = (__bf16)p0;
            Plds[w][r + 8 * half][16 + mloc] = (__bf16)p1;
        }
        __builtin_amdgcn_wave_barrier();

        const __bf16* pr = &Plds[w][mloc][0];
        const v16bf pa = mkf(ld4(pr + 16 * half), ld4(pr + 16 * half + 8));
        o0 = WMMA_BF16(pa, mkf(v0l, v0h), o0);
        o1 = WMMA_BF16(pa, mkf(v1l, v1h), o1);
        o2 = WMMA_BF16(pa, mkf(v2l, v2h), o2);
        o3 = WMMA_BF16(pa, mkf(v3l, v3h), o3);

        ka0l = nka0l; ka0h = nka0h; kb0l = nkb0l; kb0h = nkb0h;
        ka1l = nka1l; ka1h = nka1h; kb1l = nkb1l; kb1h = nkb1h;
        v0l = nv0l; v0h = nv0h; v1l = nv1l; v1h = nv1h;
        v2l = nv2l; v2h = nv2h; v3l = nv3l; v3h = nv3h;
        km0b = nkm0; km1b = nkm1;
    }

    v8f outs[4] = {o0, o1, o2, o3};
#pragma unroll
    for (int r = 0; r < 8; ++r) {
        const float inv = 1.0f / lr[r];
        const int row = qt * 16 + r + 8 * half;
        const size_t base = (size_t)(b * 2048 + row) * 1024 + h * 64 + mloc;
#pragma unroll
        for (int i = 0; i < 4; ++i)
            Aout[base + 16 * i] = (__bf16)(outs[i][r] * inv);
    }
}

// ---- kernel 3: output projection [4096,1024] x Wproj^T[1024,1024] -> f32 ---
__global__ void gemm_proj_kernel(const __bf16* __restrict__ A,
                                 const __bf16* __restrict__ Wbf,
                                 float* __restrict__ out) {
    __shared__ __bf16 lds[4][2][TILE_ELEMS];
    const int wave = blockIdx.x * (blockDim.x >> 5) + (threadIdx.x >> 5);
    const int nt = wave % 16;   // 1024/64
    const int mt = wave / 16;   // 4096/16
    const int lane = threadIdx.x & 31;
    const int mloc = lane & 15, half = lane >> 4;
    const int w = threadIdx.x >> 5;
    const int r4 = lane >> 2, s4 = lane & 3;

    const __bf16* gA0 = A + (size_t)(mt * 16 + r4) * 1024 + s4 * 8;
    const __bf16* gB0 = Wbf + (size_t)(nt * 64 + r4) * 1024 + s4 * 8;
    __bf16* buf0 = &lds[w][0][0];
    __bf16* buf1 = &lds[w][1][0];

    issue_tile(gA0, gB0, buf0, 0,  r4, s4);
    issue_tile(gA0, gB0, buf1, 32, r4, s4);

    v8f acc0 = {}, acc1 = {}, acc2 = {}, acc3 = {};
    for (int step = 0; step < 32; ++step) {
        wait_async_le<10>();
        __bf16* lb = (step & 1) ? buf1 : buf0;
        const __bf16* Ab = lb;
        const __bf16* Bb = lb + B_OFF;

        const v16bf a = mkf(ld4(Ab + mloc * 32 + 16 * half),
                            ld4(Ab + mloc * 32 + 16 * half + 8));
        const v16bf b0 = mkf(ld4(Bb + (0  + mloc) * 32 + 16 * half),
                             ld4(Bb + (0  + mloc) * 32 + 16 * half + 8));
        const v16bf b1 = mkf(ld4(Bb + (16 + mloc) * 32 + 16 * half),
                             ld4(Bb + (16 + mloc) * 32 + 16 * half + 8));
        const v16bf b2 = mkf(ld4(Bb + (32 + mloc) * 32 + 16 * half),
                             ld4(Bb + (32 + mloc) * 32 + 16 * half + 8));
        const v16bf b3 = mkf(ld4(Bb + (48 + mloc) * 32 + 16 * half),
                             ld4(Bb + (48 + mloc) * 32 + 16 * half + 8));
        acc0 = WMMA_BF16(a, b0, acc0);
        acc1 = WMMA_BF16(a, b1, acc1);
        acc2 = WMMA_BF16(a, b2, acc2);
        acc3 = WMMA_BF16(a, b3, acc3);

        wait_ds0();
        issue_tile(gA0, gB0, lb, ((step + 2) & 31) * 32, r4, s4);
    }

    v8f accs[4] = {acc0, acc1, acc2, acc3};
#pragma unroll
    for (int i = 0; i < 4; ++i) {
#pragma unroll
        for (int r = 0; r < 8; ++r) {
            const int m = mt * 16 + r + 8 * half;
            const int n = nt * 64 + 16 * i + mloc;
            out[(size_t)m * 1024 + n] = accs[i][r];
        }
    }
}

extern "C" void kernel_launch(void* const* d_in, const int* in_sizes, int n_in,
                              void* d_out, int out_size, void* d_ws, size_t ws_size,
                              hipStream_t stream) {
    const float* x = (const float*)d_in[0];
    const unsigned char* mask = (const unsigned char*)d_in[1];   // jnp bool = 1 byte
    const float* Wqkv = (const float*)d_in[2];
    const float* Wproj = (const float*)d_in[3];
    float* out = (float*)d_out;

    char* ws = (char*)d_ws;
    const size_t MiB = 1048576;
    __bf16* xbf  = (__bf16*)(ws);                  // 8 MiB
    __bf16* Wqbf = (__bf16*)(ws + 8  * MiB);       // 6 MiB
    __bf16* Wpbf = (__bf16*)(ws + 14 * MiB);       // 2 MiB
    __bf16* Qw   = (__bf16*)(ws + 16 * MiB);       // 8 MiB
    __bf16* Kw   = (__bf16*)(ws + 24 * MiB);       // 8 MiB
    __bf16* Vtw  = (__bf16*)(ws + 32 * MiB);       // 8 MiB
    __bf16* Aatt = (__bf16*)(ws + 40 * MiB);       // 8 MiB

    cvt_bf16_kernel<<<4194304 / 8 / 256, 256, 0, stream>>>(x, xbf);
    cvt_bf16_kernel<<<3145728 / 8 / 256, 256, 0, stream>>>(Wqkv, Wqbf);
    cvt_bf16_kernel<<<1048576 / 8 / 256, 256, 0, stream>>>(Wproj, Wpbf);

    gemm_qkv_kernel<<<3072, 128, 0, stream>>>(xbf, Wqbf, Qw, Kw, Vtw);
    attn_kernel<<<1024, 128, 0, stream>>>(Qw, Kw, Vtw, mask, Aatt);
    gemm_proj_kernel<<<1024, 128, 0, stream>>>(Aatt, Wpbf, out);
}
